// BaseSelfAttentionBlock_49701361549881
// MI455X (gfx1250) — matrix operand order repaired and checked by reading, not compiled, not run on hardware
//
#include <hip/hip_runtime.h>
#include <hip/hip_bf16.h>

// ---------------------------------------------------------------------------
// Transformer block (pre-LN attention + MLP) for gfx1250 / MI455X.
// All GEMMs use v_wmma_f32_16x16x32_f16 (wave32 WMMA), f16 activations,
// f32 accumulation, f32 residual/LN paths.
// GEMMs and attention both compute transposed C tiles so that each lane's
// accumulator registers hold contiguous output elements -> packed stores.
// Attention K-tiles are staged with async global->LDS copies (ASYNCcnt).
// ---------------------------------------------------------------------------

typedef __attribute__((ext_vector_type(16))) _Float16 v16h;
typedef __attribute__((ext_vector_type(8)))  _Float16 v8h;
typedef __attribute__((ext_vector_type(4)))  _Float16 v4h;
typedef __attribute__((ext_vector_type(8)))  float    v8f;

#define WMMA_F16(a, b, c) \
  __builtin_amdgcn_wmma_f32_16x16x32_f16(false, (a), false, (b), (short)0, (c), false, false)

// Load a 16x32 f16 A-fragment from a row-major matrix.
// Layout: lane l=L%16 holds row (row0+l); halves 0..15 = K hi*16+{0..15}
// (contiguous, matching the B-fragments).  -> two contiguous 16B loads per lane.
__device__ __forceinline__ v16h load_a_frag(const _Float16* base, int lda, int l, int hi) {
  const _Float16* p = base + (size_t)l * lda + hi * 16;
  v8h p0 = *(const v8h*)(p);
  v8h p1 = *(const v8h*)(p + 8);
  return __builtin_shufflevector(p0, p1, 0, 1, 2, 3, 4, 5, 6, 7, 8, 9, 10, 11, 12, 13, 14, 15);
}

// Async global->LDS copy of 8 bytes (GLOBAL_LOAD_ASYNC_TO_LDS_B64, GV mode).
// LDS destination is a byte offset in a VGPR; global source is a 64-bit VGPR address.
__device__ __forceinline__ void async_copy_b64(unsigned lds_byte_off, const void* gsrc) {
  asm volatile("global_load_async_to_lds_b64 %0, %1, off"
               :: "v"(lds_byte_off), "v"((unsigned long long)(size_t)gsrc)
               : "memory");
}
__device__ __forceinline__ void wait_asynccnt0() {
  asm volatile("s_wait_asynccnt 0x0" ::: "memory");
}

// ---------------------------------------------------------------------------
// Weight convert: W (K x N, f32, row-major) -> WT (N x K, f16, row-major).
// ---------------------------------------------------------------------------
__global__ void wcvt_kernel(const float* __restrict__ W, _Float16* __restrict__ WT,
                            int K, int N) {
  size_t i = (size_t)blockIdx.x * 256 + threadIdx.x;
  if (i >= (size_t)K * N) return;
  int n = (int)(i / K);
  int k = (int)(i % K);
  WT[i] = (_Float16)W[(size_t)k * N + n];
}

// ---------------------------------------------------------------------------
// LayerNorm over rows of 256; one wave per row; writes f32 + f16 copies.
// ---------------------------------------------------------------------------
__global__ __launch_bounds__(256) void ln_kernel(const float* __restrict__ x,
                                                 const float* __restrict__ w,
                                                 const float* __restrict__ b,
                                                 float* __restrict__ xn,
                                                 _Float16* __restrict__ xnh) {
  const int wv = threadIdx.x >> 5, lane = threadIdx.x & 31;
  const size_t row = (size_t)blockIdx.x * 8 + wv;
  const float* px = x + row * 256 + lane * 8;
  float v[8];
  float4 a0 = *(const float4*)(px);
  float4 a1 = *(const float4*)(px + 4);
  v[0] = a0.x; v[1] = a0.y; v[2] = a0.z; v[3] = a0.w;
  v[4] = a1.x; v[5] = a1.y; v[6] = a1.z; v[7] = a1.w;
  float s = 0.f, ss = 0.f;
#pragma unroll
  for (int i = 0; i < 8; ++i) { s += v[i]; ss += v[i] * v[i]; }
#pragma unroll
  for (int off = 16; off >= 1; off >>= 1) {
    s  += __shfl_xor(s, off, 32);
    ss += __shfl_xor(ss, off, 32);
  }
  float mean = s * (1.0f / 256.0f);
  float var  = ss * (1.0f / 256.0f) - mean * mean;
  float rs   = rsqrtf(var + 1e-5f);
#pragma unroll
  for (int i = 0; i < 8; ++i) {
    int c = lane * 8 + i;
    float o = (v[i] - mean) * rs * w[c] + b[c];
    xn[row * 256 + c]  = o;
    xnh[row * 256 + c] = (_Float16)o;
  }
}

// ---------------------------------------------------------------------------
// Transposed WMMA GEMM:  C^T(feature x token) = WT_tile @ Act^T.
//   A-frag = WT rows (features), B-frag = Act[token][k] (contiguous v16h).
// Lane's 8 accumulator elements = 8 contiguous features of ONE token
// -> fully packed epilogue (16B f16 stores / float4 f32 residual RMW).
// Block = 256 threads = 8 waves (2 feature x 4 token); wave tile = 32f x 64t.
// ---------------------------------------------------------------------------
enum { EPI_H16 = 0, EPI_RESF32 = 1, EPI_GELU = 2 };

template <int EPI>
__global__ __launch_bounds__(256) void gemm_wmma(const _Float16* __restrict__ Act,
                                                 const _Float16* __restrict__ WT,
                                                 const float* __restrict__ bias,
                                                 const float* __restrict__ resid,
                                                 float* __restrict__ outf,
                                                 _Float16* __restrict__ outh,
                                                 int M, int N, int K) {
  const int tid = threadIdx.x, wv = tid >> 5, lane = tid & 31;
  const int l = lane & 15, hi = lane >> 4;
  const int tw = wv & 3, fw = wv >> 2;
  const size_t tok0 = (size_t)blockIdx.x * 256 + tw * 64;
  const int feat0 = blockIdx.y * 64 + fw * 32;
  (void)M;

  v8f zero = {};
  v8f acc[2][4];  // [fi][ti]
#pragma unroll
  for (int fi = 0; fi < 2; ++fi)
#pragma unroll
    for (int ti = 0; ti < 4; ++ti) acc[fi][ti] = zero;

#pragma unroll 2
  for (int k0 = 0; k0 < K; k0 += 32) {
    v16h afr[2], bfr[4];
#pragma unroll
    for (int fi = 0; fi < 2; ++fi)
      afr[fi] = load_a_frag(WT + (size_t)(feat0 + fi * 16) * K + k0, K, l, hi);
#pragma unroll
    for (int ti = 0; ti < 4; ++ti)
      bfr[ti] = *(const v16h*)(Act + (tok0 + ti * 16 + l) * (size_t)K + k0 + hi * 16);
#pragma unroll
    for (int fi = 0; fi < 2; ++fi)
#pragma unroll
      for (int ti = 0; ti < 4; ++ti)
        acc[fi][ti] = WMMA_F16(afr[fi], bfr[ti], acc[fi][ti]);
  }

#pragma unroll
  for (int fi = 0; fi < 2; ++fi) {
    const int f0 = feat0 + fi * 16 + hi * 8;  // 8 contiguous features
    float bb[8];
    float4 bv0 = *(const float4*)(bias + f0);
    float4 bv1 = *(const float4*)(bias + f0 + 4);
    bb[0] = bv0.x; bb[1] = bv0.y; bb[2] = bv0.z; bb[3] = bv0.w;
    bb[4] = bv1.x; bb[5] = bv1.y; bb[6] = bv1.z; bb[7] = bv1.w;
#pragma unroll
    for (int ti = 0; ti < 4; ++ti) {
      const size_t tok = tok0 + ti * 16 + l;
      const size_t idx = tok * (size_t)N + f0;
      if (EPI == EPI_H16 || EPI == EPI_GELU) {
        v8h ov;
#pragma unroll
        for (int r = 0; r < 8; ++r) {
          float v = acc[fi][ti][r] + bb[r];
          if (EPI == EPI_GELU) v = 0.5f * v * (1.0f + erff(v * 0.70710678118654752f));
          ov[r] = (_Float16)v;
        }
        *(v8h*)(outh + idx) = ov;
      } else {  // EPI_RESF32
        float4 r0 = *(const float4*)(resid + idx);
        float4 r1 = *(const float4*)(resid + idx + 4);
        float4 o0, o1;
        o0.x = r0.x + acc[fi][ti][0] + bb[0];
        o0.y = r0.y + acc[fi][ti][1] + bb[1];
        o0.z = r0.z + acc[fi][ti][2] + bb[2];
        o0.w = r0.w + acc[fi][ti][3] + bb[3];
        o1.x = r1.x + acc[fi][ti][4] + bb[4];
        o1.y = r1.y + acc[fi][ti][5] + bb[5];
        o1.z = r1.z + acc[fi][ti][6] + bb[6];
        o1.w = r1.w + acc[fi][ti][7] + bb[7];
        *(float4*)(outf + idx) = o0;
        *(float4*)(outf + idx + 4) = o1;
      }
    }
  }
}

// ---------------------------------------------------------------------------
// Flash attention, transposed-score formulation, double-buffered staging.
// One block per (b,a,head); 8 waves x 64 query rows; 32-key tiles.
//   S^T = K_tile (16k x 32d) @ Q^T (32d x 16q)   -> lane owns ONE query column
//   O^T = V^T   (16d x 32k) @ P   (32k x 16q)
// Softmax in base-2 domain (exp2); reductions in-lane except one shfl_xor(16).
// K tiles: async global->LDS DMA (ASYNCcnt); V tiles: register-staged
// transpose.  Next tile's copies issue while current tile computes.
// ---------------------------------------------------------------------------
__global__ __launch_bounds__(256) void attn_flash(const _Float16* __restrict__ qkv,
                                                  _Float16* __restrict__ y) {
  __shared__ alignas(32) _Float16 sK[2][32 * 32];    // [buf][key][d]
  __shared__ alignas(32) _Float16 sVT[2][32 * 32];   // [buf][d][key]
  __shared__ alignas(32) _Float16 sP[8][64 * 32];    // per-wave [q][key]

  const int ba = blockIdx.x >> 3;  // / H
  const int h  = blockIdx.x & 7;   // % H
  const int tid = threadIdx.x, wv = tid >> 5, lane = tid & 31;
  const int l = lane & 15, hi = lane >> 4;
  const size_t tbase = (size_t)ba * 512;
  const _Float16* qb = qkv + tbase * 768 + h * 32;
  const _Float16* kb = qb + 256;
  const _Float16* vb = qb + 512;
  const int q0 = wv * 64;

  // Q^T B-fragments, pre-scaled by log2(e)/sqrt(Dh) (base-2 softmax domain).
  const _Float16 qscale = (_Float16)(0.17677669529663687f * 1.4426950408889634f);
  v16h qfr[4];
#pragma unroll
  for (int qi = 0; qi < 4; ++qi) {
    v16h t = *(const v16h*)(qb + (size_t)(q0 + qi * 16 + l) * 768 + hi * 16);
    qfr[qi] = t * qscale;
  }

  v8f zero = {};
  v8f acc[2][4];          // O^T accumulators [di][qi]
  float mrun[4], lrun[4]; // per-lane: stats of THIS lane's query column
#pragma unroll
  for (int qi = 0; qi < 4; ++qi) {
    acc[0][qi] = zero; acc[1][qi] = zero;
    mrun[qi] = -1e30f; lrun[qi] = 0.f;
  }

  _Float16* myP = &sP[wv][0];
  const int key = tid >> 3;
  const int d0  = (tid & 7) * 4;

  // LDS aperture maps addr[31:0] -> LDS offset; per-thread K dest slice.
  const unsigned ldsK = (unsigned)(size_t)(&sK[0][0]) +
                        (unsigned)key * 64u + (unsigned)(tid & 7) * 8u;

  // Prologue: async-copy K tile 0 into sK[0]; fetch V tile 0 into registers.
  async_copy_b64(ldsK, kb + (size_t)key * 768 + d0);
  v4h vv = *(const v4h*)(vb + (size_t)key * 768 + d0);

#pragma unroll 2
  for (int it = 0; it < 16; ++it) {
    const int buf = it & 1;
    // Stage current V tile (transposed) into this iteration's LDS buffer.
    sVT[buf][(d0 + 0) * 32 + key] = vv[0];
    sVT[buf][(d0 + 1) * 32 + key] = vv[1];
    sVT[buf][(d0 + 2) * 32 + key] = vv[2];
    sVT[buf][(d0 + 3) * 32 + key] = vv[3];
    wait_asynccnt0();   // this wave's K slice for tile `it` has landed
    __syncthreads();    // all waves' K/V slices visible

    // Issue next tile's copies while this one computes.
    if (it < 15) {
      async_copy_b64(ldsK + (unsigned)(buf ^ 1) * 2048u,
                     kb + (size_t)((it + 1) * 32 + key) * 768 + d0);
      vv = *(const v4h*)(vb + (size_t)((it + 1) * 32 + key) * 768 + d0);
    }

    // A-fragments: K tiles (row=key, k=d) and V^T tiles (row=d, k=key).
    v16h kfr[2], vtfr[2];
#pragma unroll
    for (int t2 = 0; t2 < 2; ++t2) {
      kfr[t2]  = load_a_frag(&sK[buf][0]  + t2 * 16 * 32, 32, l, hi);
      vtfr[t2] = load_a_frag(&sVT[buf][0] + t2 * 16 * 32, 32, l, hi);
    }

#pragma unroll
    for (int qi = 0; qi < 4; ++qi) {
      // S^T tiles (base-2 logits): rows = keys, col = this lane's query.
      v8f s0f = WMMA_F16(kfr[0], qfr[qi], zero);
      v8f s1f = WMMA_F16(kfr[1], qfr[qi], zero);

      float mx = fmaxf(s0f[0], s1f[0]);
#pragma unroll
      for (int r = 1; r < 8; ++r) mx = fmaxf(mx, fmaxf(s0f[r], s1f[r]));
      mx = fmaxf(mx, __shfl_xor(mx, 16, 32));

      float mold = mrun[qi];
      float mnew = fmaxf(mold, mx);
      float corr = exp2f(mold - mnew);

      float p0[8], p1[8], rsum = 0.f;
#pragma unroll
      for (int r = 0; r < 8; ++r) {
        p0[r] = exp2f(s0f[r] - mnew);
        p1[r] = exp2f(s1f[r] - mnew);
        rsum += p0[r] + p1[r];
      }
      rsum += __shfl_xor(rsum, 16, 32);

      mrun[qi] = mnew;
      lrun[qi] = lrun[qi] * corr + rsum;
      acc[0][qi] = acc[0][qi] * corr;
      acc[1][qi] = acc[1][qi] * corr;

      // P relayout via LDS: two packed 16B ds_store_b128.
      v8h pk0, pk1;
#pragma unroll
      for (int r = 0; r < 8; ++r) {
        pk0[r] = (_Float16)p0[r];
        pk1[r] = (_Float16)p1[r];
      }
      *(v8h*)(myP + (qi * 16 + l) * 32 +      hi * 8) = pk0;
      *(v8h*)(myP + (qi * 16 + l) * 32 + 16 + hi * 8) = pk1;

      // B-fragment of P: b[j] = P[key=hi*16+j][q=qi*16+l] -> contiguous v16h.
      v16h pfr = *(const v16h*)(myP + (qi * 16 + l) * 32 + hi * 16);
      acc[0][qi] = WMMA_F16(vtfr[0], pfr, acc[0][qi]);
      acc[1][qi] = WMMA_F16(vtfr[1], pfr, acc[1][qi]);
    }
  }

  // O^T epilogue: 8 contiguous d-elements -> one packed 16B store per (qi,di).
#pragma unroll
  for (int qi = 0; qi < 4; ++qi) {
    float inv = 1.0f / lrun[qi];
    size_t row = tbase + q0 + qi * 16 + l;
#pragma unroll
    for (int di = 0; di < 2; ++di) {
      v8h ov;
#pragma unroll
      for (int r = 0; r < 8; ++r) ov[r] = (_Float16)(acc[di][qi][r] * inv);
      *(v8h*)(y + row * 256 + h * 32 + di * 16 + hi * 8) = ov;
    }
  }
}

// ---------------------------------------------------------------------------
// Host-side orchestration.
// ---------------------------------------------------------------------------
extern "C" void kernel_launch(void* const* d_in, const int* in_sizes, int n_in,
                              void* d_out, int out_size, void* d_ws, size_t ws_size,
                              hipStream_t stream) {
  (void)in_sizes; (void)n_in; (void)out_size; (void)ws_size;
  const float* x      = (const float*)d_in[0];
  const float* ln1_w  = (const float*)d_in[1];
  const float* ln1_b  = (const float*)d_in[2];
  const float* qkv_w  = (const float*)d_in[3];
  const float* qkv_b  = (const float*)d_in[4];
  const float* proj_w = (const float*)d_in[5];
  const float* proj_b = (const float*)d_in[6];
  const float* ln2_w  = (const float*)d_in[7];
  const float* ln2_b  = (const float*)d_in[8];
  const float* fc_w   = (const float*)d_in[9];
  const float* fc_b   = (const float*)d_in[10];
  const float* fc2_w  = (const float*)d_in[11];
  const float* fc2_b  = (const float*)d_in[12];
  float* out = (float*)d_out;

  const size_t NT = 65536;  // B*A*T
  char* ws = (char*)d_ws;
  size_t o = 0;
  float*    xn   = (float*)(ws + o);    o += NT * 256 * 4;   // xn, later x2 in-place
  _Float16* xnh  = (_Float16*)(ws + o); o += NT * 256 * 2;   // xn f16, later x2n f16
  _Float16* qkvh = (_Float16*)(ws + o); o += NT * 768 * 2;   // qkv f16
  _Float16* yh   = (_Float16*)(ws + o); o += NT * 256 * 2;   // attn out f16
  float*    x2n  = (float*)(ws + o);    o += NT * 256 * 4;   // LN2 output f32
  _Float16* wq   = (_Float16*)(ws + o); o += 768 * 256 * 2;
  _Float16* wp   = (_Float16*)(ws + o); o += 256 * 256 * 2;
  _Float16* wf   = (_Float16*)(ws + o); o += 256 * 1024 * 2;
  _Float16* wf2  = (_Float16*)(ws + o); o += 1024 * 256 * 2;
  _Float16* hh   = qkvh;  // FC activations alias qkv+y region (exactly NT*1024*2 B)

  // Weight convert + transpose to f16 (N x K).
  wcvt_kernel<<<(768 * 256) / 256, 256, 0, stream>>>(qkv_w, wq, 256, 768);
  wcvt_kernel<<<(256 * 256) / 256, 256, 0, stream>>>(proj_w, wp, 256, 256);
  wcvt_kernel<<<(256 * 1024) / 256, 256, 0, stream>>>(fc_w, wf, 256, 1024);
  wcvt_kernel<<<(1024 * 256) / 256, 256, 0, stream>>>(fc2_w, wf2, 1024, 256);

  // xn = LN1(x)
  ln_kernel<<<NT / 8, 256, 0, stream>>>(x, ln1_w, ln1_b, xn, xnh);
  // qkv = xn @ qkv_w + b
  gemm_wmma<EPI_H16><<<dim3((unsigned)(NT / 256), 768 / 64), 256, 0, stream>>>(
      xnh, wq, qkv_b, nullptr, nullptr, qkvh, (int)NT, 768, 256);
  // y = attention(q,k,v)
  attn_flash<<<8 * 16 * 8, 256, 0, stream>>>(qkvh, yh);
  // x2 = xn + y @ proj_w + b   (in-place into xn buffer)
  gemm_wmma<EPI_RESF32><<<dim3((unsigned)(NT / 256), 256 / 64), 256, 0, stream>>>(
      yh, wp, proj_b, xn, xn, nullptr, (int)NT, 256, 256);
  // x2n = LN2(x2)
  ln_kernel<<<NT / 8, 256, 0, stream>>>(xn, ln2_w, ln2_b, x2n, xnh);
  // h = gelu(x2n @ fc_w + b)
  gemm_wmma<EPI_GELU><<<dim3((unsigned)(NT / 256), 1024 / 64), 256, 0, stream>>>(
      xnh, wf, fc_b, nullptr, nullptr, hh, (int)NT, 1024, 256);
  // out = x2n + h @ fc2_w + b
  gemm_wmma<EPI_RESF32><<<dim3((unsigned)(NT / 256), 256 / 64), 256, 0, stream>>>(
      hh, wf2, fc2_b, x2n, out, nullptr, (int)NT, 256, 1024);
}
